// BasicBlock_1881195675661
// MI455X (gfx1250) — matrix operand order repaired and checked by reading, not compiled, not run on hardware
//
#include <hip/hip_runtime.h>
#include <hip/hip_bf16.h>

// ---------------------------------------------------------------------------
// APR conv (3-stencil gather conv) + GroupNorm + ReLU for MI455X (gfx1250).
// Core engine: v_wmma_f32_16x16x32_bf16, wave32.
//   D(16 out-ch x 16 particles) = A(16x32 weights) x B(32x16 gathered x) + C
//   K dim = 2 taps x 16 input channels; 27 taps -> 14 pairs (last zero-padded).
// Each wave processes TWO 16-particle tiles so every LDS A-fragment read
// feeds two WMMAs (1 ds_read_b128 per WMMA).
// Weight staging global->LDS uses GLOBAL_LOAD_ASYNC_TO_LDS_B128 (ASYNCcnt
// path). GroupNorm statistics use a fully deterministic reduction
// (shfl butterflies + fixed-order sums), no float atomics anywhere.
// ---------------------------------------------------------------------------

typedef __attribute__((ext_vector_type(16))) __bf16 v16bf;
typedef __attribute__((ext_vector_type(8)))  float  v8f;
typedef __attribute__((ext_vector_type(4)))  int    v4i;

typedef __attribute__((address_space(1))) v4i* as1_v4i_ptr;  // global int4*
typedef __attribute__((address_space(3))) v4i* as3_v4i_ptr;  // LDS int4*

#define CIN 16
#define COUT 32
#define KTAP 27
#define NSTEN 3
#define NPAIR 14                        // ceil(27/2)
#define NCHUNK (NSTEN * NPAIR * 2)      // 84 A-fragments (s, pair, half)
#define WPACK_ELEMS (NCHUNK * 32 * 16)  // 43008 bf16 elements
#define WPACK_BYTES (WPACK_ELEMS * 2)   // 86016 bytes (LDS staging)
#define WAVES_PER_BLOCK 16              // 512 threads

#if __has_builtin(__builtin_amdgcn_global_load_async_to_lds_b128)
#define HAVE_ASYNC_LDS 1
#else
#define HAVE_ASYNC_LDS 0
#endif

__device__ __forceinline__ unsigned short f2bf_rne(float f) {
    union { float f; unsigned int u; } x; x.f = f;
    unsigned int u = x.u;
    unsigned int r = (u + 0x7FFFu + ((u >> 16) & 1u)) >> 16;  // round-nearest-even
    return (unsigned short)r;
}

__device__ __forceinline__ v8f zero8() {
    v8f z;
#pragma unroll
    for (int i = 0; i < 8; ++i) z[i] = 0.0f;
    return z;
}

// --- pack weights (S,Cout,Cin,K) fp32 -> bf16 as A fragments, K contiguous per lane half ---
// chunk cidx = s*28 + pair*2 + half ; per chunk: 32 lanes x 16 bf16 elements
// lane L (0-15):  M=L,    K: e -> 0..15
// lane L (16-31): M=L-16, K: e -> 16..31
// K < 16 -> tap 2*pair, channel K ; K >= 16 -> tap 2*pair+1, channel K-16
__global__ void pack_weights_kernel(const float* __restrict__ w,
                                    unsigned short* __restrict__ packed) {
    int i = blockIdx.x * blockDim.x + threadIdx.x;
    if (i >= WPACK_ELEMS) return;
    int e    = i & 15;
    int L    = (i >> 4) & 31;
    int cidx = i >> 9;
    int half = cidx & 1;
    int pair = (cidx >> 1) % NPAIR;
    int s    = cidx / (NPAIR * 2);
    int m     = L & 15;
    int chOff = (L < 16) ? 0 : 16;
    int K     = chOff + e;
    int tap   = 2 * pair + (K >= 16 ? 1 : 0);
    int c     = K & 15;
    float val = 0.0f;
    if (tap < KTAP)
        val = w[(((size_t)s * COUT + (half * 16 + m)) * CIN + c) * KTAP + tap];
    packed[i] = f2bf_rne(val);
}

// --- transpose x (C,N) fp32 -> xT (N,C) bf16: one 32B row per particle -----
__global__ void transpose_x_kernel(const float* __restrict__ x,
                                   unsigned short* __restrict__ xT, int N) {
    int n = blockIdx.x * blockDim.x + threadIdx.x;
    if (n >= N) return;
    union { unsigned short h[16]; uint4 q[2]; } row;
#pragma unroll
    for (int c = 0; c < CIN; ++c)
        row.h[c] = f2bf_rne(x[(size_t)c * N + n]);
    uint4* dst = (uint4*)(xT + (size_t)n * CIN);
    dst[0] = row.q[0];
    dst[1] = row.q[1];
}

// --- main conv kernel: one wave per 32 particles (2 x 16-particle tiles) ----
__global__ void apr_conv_wmma_kernel(const unsigned short* __restrict__ xT,
                                     const unsigned short* __restrict__ wpacked,
                                     const int* __restrict__ nbr,
                                     const int* __restrict__ lvl,
                                     float* __restrict__ y,
                                     float* __restrict__ gpartial,
                                     int N, int ngroups) {
    extern __shared__ __align__(16) unsigned short lds_w[];   // 86016 B
    __shared__ float s_part[WAVES_PER_BLOCK][16];             // per-wave stats

    const int tid  = threadIdx.x;
    const int wave = tid >> 5;
    const int lane = tid & 31;

    // cooperative staging of packed weights into LDS (5376 x b128)
    {
        const uint4* srcw = (const uint4*)wpacked;
#if HAVE_ASYNC_LDS
        uint4* ldst = (uint4*)lds_w;
        for (int i = tid; i < WPACK_BYTES / 16; i += blockDim.x) {
            __builtin_amdgcn_global_load_async_to_lds_b128(
                (as1_v4i_ptr)(srcw + i), (as3_v4i_ptr)(ldst + i), 0, 0);
        }
#if __has_builtin(__builtin_amdgcn_s_wait_asynccnt)
        __builtin_amdgcn_s_wait_asynccnt(0);
#else
        asm volatile("s_wait_asynccnt 0x0" ::: "memory");
#endif
#else
        uint4* d = (uint4*)lds_w;
        for (int i = tid; i < WPACK_BYTES / 16; i += blockDim.x) d[i] = srcw[i];
#endif
    }
    // zero this wave's stats row (inactive waves then contribute exact zeros)
    if (lane == 0) {
#pragma unroll
        for (int i = 0; i < 16; ++i) s_part[wave][i] = 0.0f;
    }
    __syncthreads();

    const int group = blockIdx.x * (blockDim.x >> 5) + wave;  // 32 particles
    const bool active = (group < ngroups);

    if (active) {
        const int lane15 = lane & 15;
        const int tapSel = lane >> 4;   // lanes 0-15: even tap, 16-31: odd tap
        int n[2];
        const int* nrow[2];
#pragma unroll
        for (int t = 0; t < 2; ++t) {
            n[t]    = group * 32 + t * 16 + lane15;
            nrow[t] = nbr + (size_t)n[t] * KTAP;
        }

        v8f acc[NSTEN][2][2];
#pragma unroll
        for (int s = 0; s < NSTEN; ++s)
#pragma unroll
            for (int h = 0; h < 2; ++h)
#pragma unroll
                for (int t = 0; t < 2; ++t) acc[s][h][t] = zero8();

#pragma unroll
        for (int pair = 0; pair < NPAIR; ++pair) {
            if (pair + 1 < NPAIR) {   // prefetch next pair's neighbor indices
                __builtin_prefetch(nrow[0] + 2 * (pair + 1), 0, 1);
                __builtin_prefetch(nrow[1] + 2 * (pair + 1), 0, 1);
            }

            // B fragments (32x16 bf16): lane handles tap (2*pair + tapSel),
            // all 16 input channels = one 32B row of xT.
            int ktap = 2 * pair + tapSel;
            if (ktap > KTAP - 1) ktap = KTAP - 1;  // tap 27: A is zero, B don't-care
            union { v16bf v; uint4 q[2]; } B[2];
#pragma unroll
            for (int t = 0; t < 2; ++t) {
                const int j = nrow[t][ktap];
                const uint4* bsrc = (const uint4*)(xT + (size_t)j * CIN);
                B[t].q[0] = bsrc[0];
                B[t].q[1] = bsrc[1];
            }

#pragma unroll
            for (int s = 0; s < NSTEN; ++s) {
#pragma unroll
                for (int h = 0; h < 2; ++h) {
                    const int cidx = (s * NPAIR + pair) * 2 + h;
                    union { v16bf v; uint4 q[2]; } A;
                    const uint4* asrc =
                        (const uint4*)lds_w + ((size_t)cidx * 32 + lane) * 2;
                    A.q[0] = asrc[0];
                    A.q[1] = asrc[1];
#pragma unroll
                    for (int t = 0; t < 2; ++t)
                        acc[s][h][t] = __builtin_amdgcn_wmma_f32_16x16x32_bf16(
                            false, A.v, false, B[t].v, (short)0, acc[s][h][t],
                            false, false);
                }
            }
        }

        // per-particle stencil select, store y, local group stats
        const int mbase = (lane < 16) ? 0 : 8;  // C/D layout: lanes 16-31 hold M=8..15
        const int gbase = mbase >> 2;           // 0 or 2
        float gsum[4] = {0.f, 0.f, 0.f, 0.f};
        float gsq[4]  = {0.f, 0.f, 0.f, 0.f};

#pragma unroll
        for (int t = 0; t < 2; ++t) {
            const int sdel = lvl[n[t]];
#pragma unroll
            for (int h = 0; h < 2; ++h) {
                v8f a0 = acc[0][h][t], a1 = acc[1][h][t], a2 = acc[2][h][t];
#pragma unroll
                for (int r = 0; r < 8; ++r) {
                    float v = (sdel == 0) ? a0[r] : ((sdel == 1) ? a1[r] : a2[r]);
                    const int o = h * 16 + mbase + r;
                    y[(size_t)o * N + n[t]] = v;
                    const int gi = h * 2 + (r >> 2);
                    gsum[gi] += v;
                    gsq[gi]  += v * v;
                }
            }
        }

        // deterministic butterfly reduction within each 16-lane half
#pragma unroll
        for (int gi = 0; gi < 4; ++gi) {
#pragma unroll
            for (int m = 8; m >= 1; m >>= 1) {
                gsum[gi] += __shfl_xor(gsum[gi], m, 32);
                gsq[gi]  += __shfl_xor(gsq[gi],  m, 32);
            }
        }
        // lanes 0 / 16 hold the totals of their half -> write disjoint entries
        if (lane == 0 || lane == 16) {
#pragma unroll
            for (int gi = 0; gi < 4; ++gi) {
                const int g = (gi >> 1) * 4 + gbase + (gi & 1); // global group 0..7
                s_part[wave][g]     = gsum[gi];
                s_part[wave][8 + g] = gsq[gi];
            }
        }
    }

    __syncthreads();
    // fixed-order per-block reduction -> one 16-float partial per block
    if (tid < 16) {
        float a = 0.0f;
        for (int wv = 0; wv < WAVES_PER_BLOCK; ++wv) a += s_part[wv][tid];
        gpartial[(size_t)blockIdx.x * 16 + tid] = a;
    }
}

// --- fixed-order final reduction + per-group mean / rstd -------------------
__global__ void finalize_stats_kernel(const float* __restrict__ gpartial,
                                      float* __restrict__ gmr,
                                      int nblocks, int N) {
    int g = threadIdx.x;
    if (g < 8) {
        float s = 0.0f, q = 0.0f;
        for (int b = 0; b < nblocks; ++b) {
            s += gpartial[(size_t)b * 16 + g];
            q += gpartial[(size_t)b * 16 + 8 + g];
        }
        float cnt  = 4.0f * (float)N;   // 4 channels per group * N particles
        float mean = s / cnt;
        float var  = q / cnt - mean * mean;
        gmr[g]     = mean;
        gmr[8 + g] = rsqrtf(var + 1e-5f);
    }
}

// --- normalize + affine + ReLU, in place on d_out (grid.y = channel) -------
__global__ void norm_relu_kernel(float* __restrict__ y,
                                 const float* __restrict__ gmr,
                                 const float* __restrict__ gamma,
                                 const float* __restrict__ beta, int N) {
    const int n = blockIdx.x * blockDim.x + threadIdx.x;
    if (n >= N) return;
    const int o = blockIdx.y;
    const int g = o >> 2;
    float v = y[(size_t)o * N + n];
    v = (v - gmr[g]) * gmr[8 + g] * gamma[o] + beta[o];
    y[(size_t)o * N + n] = v > 0.0f ? v : 0.0f;
}

// ---------------------------------------------------------------------------
extern "C" void kernel_launch(void* const* d_in, const int* in_sizes, int n_in,
                              void* d_out, int out_size, void* d_ws, size_t ws_size,
                              hipStream_t stream) {
    const float* x      = (const float*)d_in[0];  // (1, 16, N)
    const float* weight = (const float*)d_in[1];  // (3, 32, 16, 27)
    const float* gamma  = (const float*)d_in[2];  // (32,)
    const float* beta   = (const float*)d_in[3];  // (32,)
    const int*   nbr    = (const int*)d_in[4];    // (N, 27)
    const int*   lvl    = (const int*)d_in[5];    // (N,)
    float*       y      = (float*)d_out;          // (1, 32, N)

    const int N = in_sizes[5];                    // 1,000,000 (divisible by 32)
    const int ngroups = N / 32;                   // 32 particles per wave
    const int blocks  = (ngroups + WAVES_PER_BLOCK - 1) / WAVES_PER_BLOCK;

    // workspace layout
    char* ws = (char*)d_ws;
    float*          gmr      = (float*)(ws);                  // 16 floats
    unsigned short* wpacked  = (unsigned short*)(ws + 1024);  // 86016 B
    float*          gpartial = (float*)(ws + 131072);         // blocks*16 floats
    unsigned short* xT       = (unsigned short*)(ws + 262144);// N*16*2 B = 32 MB

    pack_weights_kernel<<<(WPACK_ELEMS + 255) / 256, 256, 0, stream>>>(weight, wpacked);
    transpose_x_kernel<<<(N + 255) / 256, 256, 0, stream>>>(x, xT, N);

    apr_conv_wmma_kernel<<<blocks, WAVES_PER_BLOCK * 32, WPACK_BYTES, stream>>>(
        xT, wpacked, nbr, lvl, y, gpartial, N, ngroups);

    finalize_stats_kernel<<<1, 8, 0, stream>>>(gpartial, gmr, blocks, N);

    dim3 ngrid((N + 255) / 256, COUT);
    norm_relu_kernel<<<ngrid, 256, 0, stream>>>(y, gmr, gamma, beta, N);
}
